// SimpleFineGrainedHead_63883343561447
// MI455X (gfx1250) — matrix operand order repaired and checked by reading, not compiled, not run on hardware
//
#include <hip/hip_runtime.h>

typedef __attribute__((ext_vector_type(16))) _Float16 v16h;
typedef __attribute__((ext_vector_type(8)))  _Float16 v8h;
typedef __attribute__((ext_vector_type(8)))  float    v8f;
typedef __attribute__((ext_vector_type(4)))  float    v4f;

#define K_DIM 768
#define E_DIM 256
#define B_DIM 64
#define V_DIM 197
#define VPAD  208   // 13 * 16
#define L_DIM 64
#define NEG_BIG (-3.0e38f)

__device__ __forceinline__ v16h cat16(v8h lo, v8h hi) {
  v16h r;
#pragma unroll
  for (int j = 0; j < 8; ++j) { r[j] = lo[j]; r[8 + j] = hi[j]; }
  return r;
}

// ---------------------------------------------------------------------------
// Transpose + convert the 768x256 fp32 weights into f16 [N=256][K=768] so that
// WMMA B-fragments load as two runs of 8 contiguous f16 per lane.
// ---------------------------------------------------------------------------
__global__ void prep_weights(const float* __restrict__ Wv,
                             const float* __restrict__ Wt,
                             const float* __restrict__ Wvt,
                             const float* __restrict__ Wtt,
                             _Float16* __restrict__ dst) {
  int e = blockIdx.x * blockDim.x + threadIdx.x;    // 0 .. 768*256-1
  int w = blockIdx.y;                               // which matrix
  const float* src = (w == 0) ? Wv : (w == 1) ? Wt : (w == 2) ? Wvt : Wtt;
  int k = e / E_DIM;
  int n = e % E_DIM;
  dst[(size_t)w * E_DIM * K_DIM + (size_t)n * K_DIM + k] = (_Float16)src[e];
}

// ---------------------------------------------------------------------------
// GEMM: out = X[M,768] @ W[768,256] + bias, via v_wmma_f32_16x16x32_f16.
// Block = 128 threads (4 waves), 16 rows per block, wave w owns N-tiles
// {w, w+4, w+8, w+12}. mode 0: fp32 store (cls). mode 1: row-L2-normalize,
// f16 store with per-batch row remap (tokens).
// ---------------------------------------------------------------------------
__global__ void proj_kernel(const float* __restrict__ X,
                            const _Float16* __restrict__ WT,   // [256][768]
                            const float* __restrict__ bias,    // [256]
                            float* __restrict__ outF32,
                            _Float16* __restrict__ outF16,
                            int mode, int srcRPB, int dstRPB) {
  const int lane = threadIdx.x & 31;
  const int wave = threadIdx.x >> 5;   // 0..3
  const int half = lane >> 4;          // 0 or 1
  const int ln   = lane & 15;
  const int rowbase = blockIdx.x * 16;

  __shared__ float ssq[4][16];
  __shared__ float scale[16];

  v8f c[4] = {};
  const float* arow = X + (size_t)(rowbase + ln) * K_DIM;

  for (int kb = 0; kb < K_DIM; kb += 32) {
    // A fragment (16x32 f16 layout): lane half selects K {0-7,16-23} vs {8-15,24-31}
    const float* ap = arow + kb + 8 * half;
    v4f x0 = *(const v4f*)(ap);
    v4f x1 = *(const v4f*)(ap + 4);
    v4f x2 = *(const v4f*)(ap + 16);
    v4f x3 = *(const v4f*)(ap + 20);
    v16h a;
#pragma unroll
    for (int j = 0; j < 4; ++j) {
      a[j]      = (_Float16)x0[j];
      a[4 + j]  = (_Float16)x1[j];
      a[8 + j]  = (_Float16)x2[j];
      a[12 + j] = (_Float16)x3[j];
    }
#pragma unroll
    for (int t = 0; t < 4; ++t) {
      const int n = (wave + 4 * t) * 16 + ln;
      const _Float16* bp = WT + (size_t)n * K_DIM + kb + 8 * half;
      v8h b0 = *(const v8h*)bp;
      v8h b1 = *(const v8h*)(bp + 16);
      c[t] = __builtin_amdgcn_wmma_f32_16x16x32_f16(
          false, a, false, cat16(b0, b1), (short)0, c[t], false, false);
    }
  }

  // bias: every element a lane holds shares the same n
#pragma unroll
  for (int t = 0; t < 4; ++t) {
    float bvv = bias[(wave + 4 * t) * 16 + ln];
#pragma unroll
    for (int i = 0; i < 8; ++i) c[t][i] += bvv;
  }

  if (mode == 0) {  // cls: fp32 row-major store
#pragma unroll
    for (int t = 0; t < 4; ++t) {
      int n = (wave + 4 * t) * 16 + ln;
#pragma unroll
      for (int i = 0; i < 8; ++i) {
        int row = rowbase + i + 8 * half;
        outF32[(size_t)row * E_DIM + n] = c[t][i];
      }
    }
    return;
  }

  // tokens: per-row sum of squares -> deterministic wave + LDS reduction
  float ss[8];
#pragma unroll
  for (int i = 0; i < 8; ++i) {
    float s = 0.f;
#pragma unroll
    for (int t = 0; t < 4; ++t) s += c[t][i] * c[t][i];
    s += __shfl_xor(s, 1);
    s += __shfl_xor(s, 2);
    s += __shfl_xor(s, 4);
    s += __shfl_xor(s, 8);
    ss[i] = s;
  }
  if (ln == 0) {
#pragma unroll
    for (int i = 0; i < 8; ++i) ssq[wave][i + 8 * half] = ss[i];
  }
  __syncthreads();
  if (threadIdx.x < 16) {
    float s = ssq[0][threadIdx.x] + ssq[1][threadIdx.x] +
              ssq[2][threadIdx.x] + ssq[3][threadIdx.x];
    scale[threadIdx.x] = 1.0f / fmaxf(sqrtf(s), 1e-12f);
  }
  __syncthreads();

#pragma unroll
  for (int t = 0; t < 4; ++t) {
    int n = (wave + 4 * t) * 16 + ln;
#pragma unroll
    for (int i = 0; i < 8; ++i) {
      int r = i + 8 * half;
      int row = rowbase + r;
      int bidx = row / srcRPB;
      int v = row - bidx * srcRPB;
      outF16[((size_t)bidx * dstRPB + v) * E_DIM + n] =
          (_Float16)(c[t][i] * scale[r]);
    }
  }
}

// ---------------------------------------------------------------------------
// sim[b,q] block: v_tok[b] (197x256) @ t_tok[q]^T (256x64), fused reductions.
// Block = 128 threads (4 waves); wave w owns t-columns [16w, 16w+16).
// B fragments (loop-invariant across M-tiles) are hoisted into 64 VGPRs.
// ---------------------------------------------------------------------------
__global__ void sim_kernel(const _Float16* __restrict__ vtok,  // [64][208][256]
                           const _Float16* __restrict__ ttok,  // [64][64][256]
                           const int* __restrict__ text_length,
                           float* __restrict__ i2t_out,
                           float* __restrict__ t2i_out) {
  const int b = blockIdx.x;
  const int q = blockIdx.y;
  const int lane = threadIdx.x & 31;
  const int wave = threadIdx.x >> 5;
  const int half = lane >> 4;
  const int ln   = lane & 15;

  const _Float16* A  = vtok + (size_t)b * VPAD * E_DIM;
  const _Float16* Bm = ttok + (size_t)q * L_DIM * E_DIM;

  __shared__ float wrmax[4][16];
  __shared__ float colmax_s[L_DIM];

  // B fragment source row: t = wave*16 + ln (t_tok row-major => contiguous in k)
  const _Float16* brow = Bm + (size_t)(wave * 16 + ln) * E_DIM;

  // Hoist the 8 B fragments (full K=256 for this wave's 16 t-columns).
  v16h bf[8];
#pragma unroll
  for (int kk = 0; kk < 8; ++kk) {
    const _Float16* bp = brow + kk * 32 + 8 * half;
    v8h b0 = *(const v8h*)bp;
    v8h b1 = *(const v8h*)(bp + 16);
    bf[kk] = cat16(b0, b1);
  }

  float i2t_acc = 0.f;       // meaningful in thread 0 only
  float cmax = NEG_BIG;      // per-lane column-max accumulator

  for (int mt = 0; mt < 13; ++mt) {
    v8f c = {};
    const _Float16* arow = A + (size_t)(mt * 16 + ln) * E_DIM;
    if (mt + 1 < 13) {  // prefetch next M-tile's A rows (global_prefetch_b8)
      __builtin_prefetch(A + (size_t)((mt + 1) * 16 + ln) * E_DIM, 0, 1);
    }
#pragma unroll
    for (int kk = 0; kk < 8; ++kk) {
      const _Float16* ap = arow + kk * 32 + 8 * half;
      v8h a0 = *(const v8h*)ap;
      v8h a1 = *(const v8h*)(ap + 16);
      c = __builtin_amdgcn_wmma_f32_16x16x32_f16(
          false, cat16(a0, a1), false, bf[kk], (short)0, c, false, false);
    }

#pragma unroll
    for (int i = 0; i < 8; ++i) {
      // row max over this wave's 16 t-columns (i2t uses max over ALL t)
      float m = c[i];
      m = fmaxf(m, __shfl_xor(m, 1));
      m = fmaxf(m, __shfl_xor(m, 2));
      m = fmaxf(m, __shfl_xor(m, 4));
      m = fmaxf(m, __shfl_xor(m, 8));
      if (ln == 0) wrmax[wave][i + 8 * half] = m;
      // column max over v: mask rows beyond 197
      int row = mt * 16 + i + 8 * half;
      float cv = (row < V_DIM) ? c[i] : NEG_BIG;
      cmax = fmaxf(cmax, cv);
    }
    __syncthreads();
    if (wave == 0) {  // combine the 4 waves' row maxima, accumulate i2t sum
      float rm = 0.f;
      if (lane < 16) {
        int row = mt * 16 + lane;
        if (row < V_DIM) {
          rm = fmaxf(fmaxf(wrmax[0][lane], wrmax[1][lane]),
                     fmaxf(wrmax[2][lane], wrmax[3][lane]));
        }
      }
      rm += __shfl_xor(rm, 1);
      rm += __shfl_xor(rm, 2);
      rm += __shfl_xor(rm, 4);
      rm += __shfl_xor(rm, 8);
      if (lane == 0) i2t_acc += rm;
    }
    __syncthreads();
  }

  // finalize column maxima: combine the two 16-lane halves
  cmax = fmaxf(cmax, __shfl_xor(cmax, 16));
  if (half == 0) colmax_s[wave * 16 + ln] = cmax;
  __syncthreads();

  if (threadIdx.x == 0) {
    int tl = text_length[b];            // NOTE: reference masks by *visual* index b
    if (tl > L_DIM) tl = L_DIM;
    if (tl < 0) tl = 0;
    float s = 0.f;
    for (int t = 0; t < tl; ++t) s += colmax_s[t];
    float denom = fmaxf((float)tl, 1e-6f);
    t2i_out[b * B_DIM + q] = s / denom;
    i2t_out[b * B_DIM + q] = i2t_acc / (float)V_DIM;
  }
}

// ---------------------------------------------------------------------------
extern "C" void kernel_launch(void* const* d_in, const int* in_sizes, int n_in,
                              void* d_out, int out_size, void* d_ws, size_t ws_size,
                              hipStream_t stream) {
  const float* visual_cls     = (const float*)d_in[0];
  const float* textual_cls    = (const float*)d_in[1];
  const float* visual_tokens  = (const float*)d_in[2];
  const float* textual_tokens = (const float*)d_in[3];
  const int*   text_length    = (const int*)d_in[4];
  const float* Wv  = (const float*)d_in[5];
  const float* bv  = (const float*)d_in[6];
  const float* Wt  = (const float*)d_in[7];
  const float* bt  = (const float*)d_in[8];
  const float* Wvt = (const float*)d_in[9];
  const float* bvt = (const float*)d_in[10];
  const float* Wtt = (const float*)d_in[11];
  const float* btt = (const float*)d_in[12];

  float* out       = (float*)d_out;
  float* v_cls_out = out;                        // 64*256
  float* t_cls_out = out + B_DIM * E_DIM;        // 64*256
  float* i2t_out   = out + 2 * B_DIM * E_DIM;    // 64*64
  float* t2i_out   = i2t_out + B_DIM * B_DIM;    // 64*64

  // Workspace layout (f16): 4 transposed weights, padded v_tok, t_tok (~10 MB)
  char* ws = (char*)d_ws;
  const size_t WSZ = (size_t)E_DIM * K_DIM;      // elements per weight matrix
  _Float16* wbuf  = (_Float16*)ws;
  _Float16* Wv_t  = wbuf + 0 * WSZ;
  _Float16* Wt_t  = wbuf + 1 * WSZ;
  _Float16* Wvt_t = wbuf + 2 * WSZ;
  _Float16* Wtt_t = wbuf + 3 * WSZ;
  _Float16* vtok  = wbuf + 4 * WSZ;                         // [64][208][256]
  _Float16* ttok  = vtok + (size_t)B_DIM * VPAD * E_DIM;    // [64][64][256]

  prep_weights<<<dim3((K_DIM * E_DIM) / 256, 4), 256, 0, stream>>>(
      Wv, Wt, Wvt, Wtt, wbuf);

  // cls projections: 64 rows each -> 4 blocks of 16 rows
  proj_kernel<<<4, 128, 0, stream>>>(visual_cls,  Wv_t, bv, v_cls_out, nullptr, 0, 1, 1);
  proj_kernel<<<4, 128, 0, stream>>>(textual_cls, Wt_t, bt, t_cls_out, nullptr, 0, 1, 1);

  // token projections + l2norm: 12608 rows and 4096 rows (both /16 exactly)
  proj_kernel<<<(B_DIM * V_DIM) / 16, 128, 0, stream>>>(
      visual_tokens, Wvt_t, bvt, nullptr, vtok, 1, V_DIM, VPAD);
  proj_kernel<<<(B_DIM * L_DIM) / 16, 128, 0, stream>>>(
      textual_tokens, Wtt_t, btt, nullptr, ttok, 1, L_DIM, L_DIM);

  // late-interaction similarity + fused max/sum reductions
  sim_kernel<<<dim3(B_DIM, B_DIM), 128, 0, stream>>>(
      vtok, ttok, text_length, i2t_out, t2i_out);
}
